// DocREModel_36867999269449
// MI455X (gfx1250) — matrix operand order repaired and checked by reading, not compiled, not run on hardware
//
#include <hip/hip_runtime.h>
#include <hip/hip_bf16.h>

typedef _Float16 f16;
typedef _Float16 v16h __attribute__((ext_vector_type(16)));
typedef _Float16 v8h  __attribute__((ext_vector_type(8)));
typedef _Float16 v4h  __attribute__((ext_vector_type(4)));
typedef float    v8f  __attribute__((ext_vector_type(8)));

// Problem constants
// B=8, C=256, CQ=32, H=W=128, HW=16384

// ---------------------------------------------------------------------------
// Kernel 0: pack Wq(32x256), Wk(32x256), Wv(256x256) into WMMA A-fragment
// order: wpack[otile][kstep][lane][e], otile in 0..19 (o = otile*16+m),
// kstep in 0..7 (c = kstep*32 + K(e,half)).
// A-layout as used here (f16 16x32, contiguous k, same as B; not the ISA table order): m = lane&15, half = lane>>4,
//   K(e) = half*16 + e
// ---------------------------------------------------------------------------
__global__ __launch_bounds__(256) void pack_weights(const float* __restrict__ Wq,
                                                    const float* __restrict__ Wk,
                                                    const float* __restrict__ Wv,
                                                    f16* __restrict__ wpack) {
    int t = blockIdx.x * 256 + threadIdx.x;         // 81920 total
    int e    = t & 15;
    int lane = (t >> 4) & 31;
    int ks   = (t >> 9) & 7;
    int ot   = t >> 12;
    if (ot >= 20) return;
    int m = lane & 15, half = lane >> 4;
    int o = ot * 16 + m;
    int K = (half << 4) + e;
    int c = ks * 32 + K;
    float wv;
    if (o < 32)       wv = Wq[o * 256 + c];
    else if (o < 64)  wv = Wk[(o - 32) * 256 + c];
    else              wv = Wv[(o - 64) * 256 + c];
    wpack[t] = (f16)wv;
}

// ---------------------------------------------------------------------------
// Kernel 1: q/k/v projection. One block = 16 pixels x all 320 out channels.
// GEMM [320x256] * [256x16] via v_wmma_f32_16x16x32_f16, K-loop of 8.
// Writes q,k,v as f16 plus transposed v (vT[b][c][w][h]) for column GEMMs.
// ---------------------------------------------------------------------------
__global__ __launch_bounds__(128) void qkv_kernel(const float* __restrict__ x,
                                                  const f16* __restrict__ wpack,
                                                  const float* __restrict__ bq,
                                                  const float* __restrict__ bk,
                                                  const float* __restrict__ bv,
                                                  f16* __restrict__ qh,
                                                  f16* __restrict__ kh,
                                                  f16* __restrict__ vh,
                                                  f16* __restrict__ vT) {
    __shared__ f16 X[16][256];                      // [pixel][channel], 8KB
    int b     = blockIdx.x >> 10;
    int pt    = blockIdx.x & 1023;
    int pbase = pt << 4;
    int tid   = threadIdx.x;

    for (int i = 0; i < 32; ++i) {
        int idx = i * 128 + tid;
        int c = idx >> 4, pp = idx & 15;
        X[pp][c] = (f16)x[(size_t)(b * 256 + c) * 16384 + pbase + pp];
    }
    __syncthreads();

    int wave = tid >> 5, lane = tid & 31;
    int nl = lane & 15, half = lane >> 4;

    for (int j = 0; j < 5; ++j) {
        int ot = wave + 4 * j;                      // 4 waves x 5 -> 20 o-tiles
        v8f acc = {};
        for (int ks = 0; ks < 8; ++ks) {
            v16h a  = *(const v16h*)(wpack + (size_t)((ot * 8 + ks) * 32 + lane) * 16);
            v16h bf = *(const v16h*)(&X[nl][ks * 32 + half * 16]);
            acc = __builtin_amdgcn_wmma_f32_16x16x32_f16(false, a, false, bf,
                                                         (short)0, acc, false, false);
        }
        int p = pbase + nl;
        for (int r = 0; r < 8; ++r) {
            int o = ot * 16 + half * 8 + r;
            float val = acc[r];
            if (o < 32) {
                val += bq[o];
                qh[(size_t)(b * 32 + o) * 16384 + p] = (f16)val;
            } else if (o < 64) {
                val += bk[o - 32];
                kh[(size_t)(b * 32 + o - 32) * 16384 + p] = (f16)val;
            } else {
                int c = o - 64;
                val += bv[c];
                f16 hv = (f16)val;
                vh[(size_t)(b * 256 + c) * 16384 + p] = hv;
                int hh = p >> 7, ww = p & 127;
                vT[(size_t)(b * 256 + c) * 16384 + ww * 128 + hh] = hv;
            }
        }
    }
}

// helper: build an A fragment from an LDS row (two contiguous 16B loads)
__device__ __forceinline__ v16h load_frag_a(const f16* row, int half) {
    v8h lo = *(const v8h*)(row + half * 16);
    v8h hi = *(const v8h*)(row + half * 16 + 8);
    v16h a;
#pragma unroll
    for (int i = 0; i < 8; ++i) { a[i] = lo[i]; a[8 + i] = hi[i]; }
    return a;
}

// ---------------------------------------------------------------------------
// Kernel 2: eH logits per (b,w) column: eH[h][g] = sum_o q[o,h,w]*k[o,g,w].
// 128x128 output, K=32 -> exactly one WMMA per 16x16 tile. Diagonal -> -inf.
// Layout eH[b][w][h][g], f16.
// ---------------------------------------------------------------------------
__global__ __launch_bounds__(256) void eh_kernel(const f16* __restrict__ qh,
                                                 const f16* __restrict__ kh,
                                                 f16* __restrict__ eH) {
    __shared__ f16 Qc[128][32];
    __shared__ f16 Kc[128][32];
    int b = blockIdx.x >> 7;
    int w = blockIdx.x & 127;
    int tid = threadIdx.x;
    for (int i = 0; i < 16; ++i) {
        int idx = i * 256 + tid;
        int h = idx >> 5, o = idx & 31;
        size_t off = (size_t)(b * 32 + o) * 16384 + h * 128 + w;
        Qc[h][o] = qh[off];
        Kc[h][o] = kh[off];
    }
    __syncthreads();

    int wave = tid >> 5, lane = tid & 31;
    int nl = lane & 15, half = lane >> 4;
    int ht = wave;                                   // 8 waves -> 8 h-tiles
    v16h a = load_frag_a(&Qc[ht * 16 + nl][0], half);
    size_t base = (size_t)(b * 128 + w) * 128 * 128;
    for (int gt = 0; gt < 8; ++gt) {
        v16h bf = *(const v16h*)(&Kc[gt * 16 + nl][half * 16]);
        v8f acc = {};
        acc = __builtin_amdgcn_wmma_f32_16x16x32_f16(false, a, false, bf,
                                                     (short)0, acc, false, false);
        for (int r = 0; r < 8; ++r) {
            int hg = ht * 16 + half * 8 + r;
            int gg = gt * 16 + nl;
            f16 outv = (hg == gg) ? (f16)(-__builtin_huge_valf()) : (f16)acc[r];
            eH[base + (size_t)hg * 128 + gg] = outv;
        }
    }
}

// ---------------------------------------------------------------------------
// Kernel 3: eW logits per (b,h) row: eW[w][g] = sum_o q[o,h,w]*k[o,h,g].
// Layout eW[b][h][w][g], f16. No mask.
// ---------------------------------------------------------------------------
__global__ __launch_bounds__(256) void ew_kernel(const f16* __restrict__ qh,
                                                 const f16* __restrict__ kh,
                                                 f16* __restrict__ eW) {
    __shared__ f16 Qr[128][32];
    __shared__ f16 Kr[128][32];
    int b = blockIdx.x >> 7;
    int h = blockIdx.x & 127;
    int tid = threadIdx.x;
    for (int i = 0; i < 16; ++i) {
        int idx = i * 256 + tid;
        int o = idx >> 7, w = idx & 127;             // w contiguous -> coalesced
        size_t off = (size_t)(b * 32 + o) * 16384 + h * 128 + w;
        Qr[w][o] = qh[off];
        Kr[w][o] = kh[off];
    }
    __syncthreads();

    int wave = tid >> 5, lane = tid & 31;
    int nl = lane & 15, half = lane >> 4;
    int wt = wave;
    v16h a = load_frag_a(&Qr[wt * 16 + nl][0], half);
    size_t base = (size_t)(b * 128 + h) * 128 * 128;
    for (int gt = 0; gt < 8; ++gt) {
        v16h bf = *(const v16h*)(&Kr[gt * 16 + nl][half * 16]);
        v8f acc = {};
        acc = __builtin_amdgcn_wmma_f32_16x16x32_f16(false, a, false, bf,
                                                     (short)0, acc, false, false);
        for (int r = 0; r < 8; ++r) {
            int wg = wt * 16 + half * 8 + r;
            int gg = gt * 16 + nl;
            eW[base + (size_t)wg * 128 + gg] = (f16)acc[r];
        }
    }
}

// ---------------------------------------------------------------------------
// Kernel 4: softmax over the concatenated 256 logits [eH row | eW row] per
// pixel. One wave32 per pixel, 8 values per lane, shuffle reductions.
// Normalized attention written back in place as f16.
// ---------------------------------------------------------------------------
__global__ __launch_bounds__(256) void softmax_kernel(f16* __restrict__ eH,
                                                      f16* __restrict__ eW) {
    int wave = threadIdx.x >> 5, lane = threadIdx.x & 31;
    int pix = blockIdx.x * 8 + wave;
    int b = pix >> 14;
    int rem = pix & 16383;
    int h = rem >> 7, w = rem & 127;
    f16* rowH = eH + (size_t)((b * 128 + w) * 128 + h) * 128;
    f16* rowW = eW + (size_t)((b * 128 + h) * 128 + w) * 128;

    v4h hv = *(const v4h*)(rowH + lane * 4);
    v4h wv = *(const v4h*)(rowW + lane * 4);
    float vals[8];
#pragma unroll
    for (int i = 0; i < 4; ++i) { vals[i] = (float)hv[i]; vals[4 + i] = (float)wv[i]; }
    float m = vals[0];
#pragma unroll
    for (int i = 1; i < 8; ++i) m = fmaxf(m, vals[i]);
    for (int off = 16; off >= 1; off >>= 1) m = fmaxf(m, __shfl_xor(m, off, 32));
    float s = 0.f;
#pragma unroll
    for (int i = 0; i < 8; ++i) { vals[i] = __expf(vals[i] - m); s += vals[i]; }
    for (int off = 16; off >= 1; off >>= 1) s += __shfl_xor(s, off, 32);
    float inv = 1.0f / s;
#pragma unroll
    for (int i = 0; i < 4; ++i) {
        hv[i] = (f16)(vals[i] * inv);
        wv[i] = (f16)(vals[4 + i] * inv);
    }
    *(v4h*)(rowH + lane * 4) = hv;
    *(v4h*)(rowW + lane * 4) = wv;
}

// ---------------------------------------------------------------------------
// Kernel 5: outH per (b,w,cblock): Out[c][h] = sum_g V[c][g]*AH[h][g]
// with V[c][g] = v[b,c,g,w] loaded from vT (contiguous in g).
// Raw f32 result into d_out.
// ---------------------------------------------------------------------------
__global__ __launch_bounds__(256) void outh_kernel(const f16* __restrict__ vT,
                                                   const f16* __restrict__ aH,
                                                   float* __restrict__ out) {
    __shared__ f16 Vc[64][128];                      // 16KB
    __shared__ f16 Ah[128][128];                     // 32KB
    int id = blockIdx.x;
    int b = id >> 9;
    int rem = id & 511;
    int w = rem >> 2;
    int cb = rem & 3;                                // 4 c-blocks of 64
    int tid = threadIdx.x;
    for (int i = 0; i < 32; ++i) {
        int idx = i * 256 + tid;
        int ci = idx >> 7, g = idx & 127;
        Vc[ci][g] = vT[(size_t)(b * 256 + cb * 64 + ci) * 16384 + w * 128 + g];
    }
    for (int i = 0; i < 64; ++i) {
        int idx = i * 256 + tid;
        int hh = idx >> 7, g = idx & 127;
        Ah[hh][g] = aH[((size_t)(b * 128 + w) * 128 + hh) * 128 + g];
    }
    __syncthreads();

    int wave = tid >> 5, lane = tid & 31;
    int nl = lane & 15, half = lane >> 4;
    int ct = wave & 3;
    int hbase = (wave >> 2) * 4;
    for (int htl = 0; htl < 4; ++htl) {
        int ht = hbase + htl;
        v8f acc = {};
        for (int ks = 0; ks < 8; ++ks) {
            v16h a  = load_frag_a(&Vc[ct * 16 + nl][ks * 32], half);
            v16h bf = *(const v16h*)(&Ah[ht * 16 + nl][ks * 32 + half * 16]);
            acc = __builtin_amdgcn_wmma_f32_16x16x32_f16(false, a, false, bf,
                                                         (short)0, acc, false, false);
        }
        for (int r = 0; r < 8; ++r) {
            int cg = cb * 64 + ct * 16 + half * 8 + r;
            int hg = ht * 16 + nl;
            out[((size_t)(b * 256 + cg) * 128 + hg) * 128 + w] = acc[r];
        }
    }
}

// ---------------------------------------------------------------------------
// Kernel 6: outW per (b,h,cblock): Out[c][w] = sum_g V[c][g]*AW[w][g]
// with V[c][g] = v[b,c,h,g] (contiguous). Fuses gamma*(outH+outW)+x,
// coalesced final stores.
// ---------------------------------------------------------------------------
__global__ __launch_bounds__(256) void outw_kernel(const f16* __restrict__ vh,
                                                   const f16* __restrict__ aW,
                                                   const float* __restrict__ x,
                                                   const float* __restrict__ gamma,
                                                   float* __restrict__ out) {
    __shared__ f16 Vr[64][128];
    __shared__ f16 Aw[128][128];
    int id = blockIdx.x;
    int b = id >> 9;
    int rem = id & 511;
    int h = rem >> 2;
    int cb = rem & 3;
    int tid = threadIdx.x;
    for (int i = 0; i < 32; ++i) {
        int idx = i * 256 + tid;
        int ci = idx >> 7, g = idx & 127;
        Vr[ci][g] = vh[(size_t)(b * 256 + cb * 64 + ci) * 16384 + h * 128 + g];
    }
    for (int i = 0; i < 64; ++i) {
        int idx = i * 256 + tid;
        int ww = idx >> 7, g = idx & 127;
        Aw[ww][g] = aW[((size_t)(b * 128 + h) * 128 + ww) * 128 + g];
    }
    __syncthreads();

    float gm = gamma[0];
    int wave = tid >> 5, lane = tid & 31;
    int nl = lane & 15, half = lane >> 4;
    int ct = wave & 3;
    int wbase = (wave >> 2) * 4;
    for (int wtl = 0; wtl < 4; ++wtl) {
        int wt = wbase + wtl;
        v8f acc = {};
        for (int ks = 0; ks < 8; ++ks) {
            v16h a  = load_frag_a(&Vr[ct * 16 + nl][ks * 32], half);
            v16h bf = *(const v16h*)(&Aw[wt * 16 + nl][ks * 32 + half * 16]);
            acc = __builtin_amdgcn_wmma_f32_16x16x32_f16(false, a, false, bf,
                                                         (short)0, acc, false, false);
        }
        for (int r = 0; r < 8; ++r) {
            int cg = cb * 64 + ct * 16 + half * 8 + r;
            int wg = wt * 16 + nl;
            size_t oidx = ((size_t)(b * 256 + cg) * 128 + h) * 128 + wg;
            out[oidx] = gm * (out[oidx] + acc[r]) + x[oidx];
        }
    }
}

// ---------------------------------------------------------------------------
extern "C" void kernel_launch(void* const* d_in, const int* in_sizes, int n_in,
                              void* d_out, int out_size, void* d_ws, size_t ws_size,
                              hipStream_t stream) {
    const float* x     = (const float*)d_in[0];
    const float* Wq    = (const float*)d_in[1];
    const float* bq    = (const float*)d_in[2];
    const float* Wk    = (const float*)d_in[3];
    const float* bk    = (const float*)d_in[4];
    const float* Wv    = (const float*)d_in[5];
    const float* bv    = (const float*)d_in[6];
    const float* gamma = (const float*)d_in[7];
    float* out = (float*)d_out;

    char* ws = (char*)d_ws;
    size_t off = 0;
    f16* qh = (f16*)(ws + off); off += (size_t)8 * 32 * 16384 * 2;      // 8MB
    f16* kh = (f16*)(ws + off); off += (size_t)8 * 32 * 16384 * 2;      // 8MB
    f16* vh = (f16*)(ws + off); off += (size_t)8 * 256 * 16384 * 2;     // 64MB
    f16* vT = (f16*)(ws + off); off += (size_t)8 * 256 * 16384 * 2;     // 64MB
    f16* eH = (f16*)(ws + off); off += (size_t)8 * 128 * 128 * 128 * 2; // 32MB
    f16* eW = (f16*)(ws + off); off += (size_t)8 * 128 * 128 * 128 * 2; // 32MB
    f16* wpack = (f16*)(ws + off); off += (size_t)81920 * 2;            // 160KB

    pack_weights<<<320, 256, 0, stream>>>(Wq, Wk, Wv, wpack);
    qkv_kernel<<<8 * 1024, 128, 0, stream>>>(x, wpack, bq, bk, bv, qh, kh, vh, vT);
    eh_kernel<<<8 * 128, 256, 0, stream>>>(qh, kh, eH);
    ew_kernel<<<8 * 128, 256, 0, stream>>>(qh, kh, eW);
    softmax_kernel<<<16384, 256, 0, stream>>>(eH, eW);
    outh_kernel<<<8 * 512, 256, 0, stream>>>(vT, eH, out);
    outw_kernel<<<8 * 512, 256, 0, stream>>>(vh, eW, x, gamma, out);
}
